// TransformerBlock_29746943492502
// MI455X (gfx1250) — matrix operand order repaired and checked by reading, not compiled, not run on hardware
//
#include <hip/hip_runtime.h>

// ---------------------------------------------------------------------------
// Transformer block (B=2, N=2048, D=768, H=12, hd=64, MLP=3072) for gfx1250.
// GEMMs + attention einsums on v_wmma_f32_16x16x32_bf16 (bf16 in, f32 acc).
// B-operand tiles staged to LDS with the Tensor Data Mover (double-buffered,
// s_wait_tensorcnt), fp32 LayerNorm / softmax / residuals.
// ---------------------------------------------------------------------------

#define DIM      768
#define HEADS    12
#define HEAD_DIM 64
#define MLP_DIM  3072
#define SEQ      2048
#define BATCH    2
#define TOKENS   (BATCH*SEQ)   /* 4096 */
#define QKV_DIM  (3*DIM)       /* 2304 */
#define LN_EPS   1e-5f

typedef __bf16 bf16_t;
typedef __attribute__((ext_vector_type(16))) __bf16        v16bf;
typedef __attribute__((ext_vector_type(8)))  float         v8f;
typedef __attribute__((ext_vector_type(4)))  unsigned int  v4u;
typedef __attribute__((ext_vector_type(8)))  int           v8i;
typedef __attribute__((ext_vector_type(4)))  int           v4i;

union AFrag { v16bf v; v4u u[2]; };

__device__ __forceinline__ bf16_t f2bf(float f) {
  unsigned u = __builtin_bit_cast(unsigned, f);
  u += 0x7FFFu + ((u >> 16) & 1u);              // round-to-nearest-even
  unsigned short h = (unsigned short)(u >> 16);
  return __builtin_bit_cast(bf16_t, h);
}

#define HAS_TDM __has_builtin(__builtin_amdgcn_tensor_load_to_lds)

#if HAS_TDM
// Issue a TDM load of a 2D bf16 tile: tile_n rows x 32 elems/row, row stride
// tensor_k elems, from gaddr into LDS offset lds_off. D# layout per
// CDNA5 ISA (group0 128b, group1 256b; groups 2/3 zero for
// a 2D tile). Tracked by TENSORcnt.
__device__ __forceinline__ void tdm_load_b_tile(const bf16_t* gaddr,
                                                unsigned lds_off,
                                                int tensor_k, int tensor_n,
                                                int tile_n) {
  const unsigned long long ga = (unsigned long long)(size_t)gaddr;
  v4u g0;
  g0.x = 1u;                                        // count=1, no gather
  g0.y = lds_off;                                   // lds_addr [63:32]
  g0.z = (unsigned)(ga & 0xFFFFFFFFu);              // global_addr [95:64]
  g0.w = (unsigned)((ga >> 32) & 0x01FFFFFFu)       // global_addr [120:96]
       | (2u << 30);                                // type=2 ("image")
  v8i g1;
  g1[0] = (1 << 16);                                // data_size=1 -> 2 bytes
  g1[1] = (tensor_k & 0xFFFF) << 16;                // tensor_dim0 [15:0]
  g1[2] = ((tensor_k >> 16) & 0xFFFF)               // tensor_dim0 [31:16]
        | ((tensor_n & 0xFFFF) << 16);              // tensor_dim1 [15:0]
  g1[3] = ((tensor_n >> 16) & 0xFFFF)               // tensor_dim1 [31:16]
        | (32 << 16);                               // tile_dim0 = 32 elems
  g1[4] = (tile_n & 0xFFFF);                        // tile_dim1; tile_dim2=0
  g1[5] = tensor_k;                                 // dim0_stride [31:0]
  g1[6] = 0;                                        // stride hi, dim1_stride lo
  g1[7] = 0;
  v4i z4 = {0, 0, 0, 0};
#if __has_include(<hip/amd_detail/amd_gfx1250_TDM.h>)
  v8i z8 = {0, 0, 0, 0, 0, 0, 0, 0};                // clang-23 6-arg form
  __builtin_amdgcn_tensor_load_to_lds(g0, g1, z4, z4, z8, 0);
#else
  __builtin_amdgcn_tensor_load_to_lds(g0, g1, z4, z4, 0);  // ROCm 7.2 5-arg
#endif
}
#endif

// ---------------------------------------------------------------------------
// Weight convert + transpose: src fp32 [K][N] row-major -> dst bf16 [N][K].
// ---------------------------------------------------------------------------
__global__ __launch_bounds__(256) void cvt_transpose(
    const float* __restrict__ src, bf16_t* __restrict__ dst, int K, int N) {
  int i = blockIdx.x * 256 + threadIdx.x;
  if (i >= K * N) return;
  int k = i / N, n = i - k * N;
  dst[(size_t)n * K + k] = f2bf(src[i]);
}

// ---------------------------------------------------------------------------
// LayerNorm (one 256-thread block per token, 768 = 3 * 256), fp32 -> bf16.
// ---------------------------------------------------------------------------
__global__ __launch_bounds__(256) void layernorm_bf16(
    const float* __restrict__ x, const float* __restrict__ w,
    const float* __restrict__ b, bf16_t* __restrict__ out) {
  const int tok = blockIdx.x;
  const float* xr = x + (size_t)tok * DIM;
  const int tid = threadIdx.x;
  float v0 = xr[tid], v1 = xr[tid + 256], v2 = xr[tid + 512];
  float s = v0 + v1 + v2;
  float q = v0 * v0 + v1 * v1 + v2 * v2;
  const int lane = tid & 31, wid = tid >> 5;
  #pragma unroll
  for (int m = 16; m; m >>= 1) {
    s += __shfl_xor(s, m, 32);
    q += __shfl_xor(q, m, 32);
  }
  __shared__ float rs[8], rq[8];
  if (lane == 0) { rs[wid] = s; rq[wid] = q; }
  __syncthreads();
  float S = 0.f, Q = 0.f;
  #pragma unroll
  for (int i = 0; i < 8; i++) { S += rs[i]; Q += rq[i]; }
  const float mu   = S * (1.0f / DIM);
  const float var  = Q * (1.0f / DIM) - mu * mu;
  const float rstd = rsqrtf(var + LN_EPS);
  bf16_t* o = out + (size_t)tok * DIM;
  o[tid      ] = f2bf((v0 - mu) * rstd * w[tid      ] + b[tid      ]);
  o[tid + 256] = f2bf((v1 - mu) * rstd * w[tid + 256] + b[tid + 256]);
  o[tid + 512] = f2bf((v2 - mu) * rstd * w[tid + 512] + b[tid + 512]);
}

// ---------------------------------------------------------------------------
// Cooperative WMMA GEMM: C[M][N] = A[M][K](bf16) * Wt[N][K](bf16) + bias.
// Block = 256 threads = 8 waves, block tile 128M x 128N, K-step 32.
// Waves arranged 4(M) x 2(N); each wave computes 32M x 64N (8 accumulators,
// A-fragment reused across 4 B tiles, B fragment reused across 2 M tiles).
// B tile (128N x 32K, 8 KB) staged to LDS by the Tensor Data Mover, double
// buffered, completion via s_wait_tensorcnt + block barrier.
// ---------------------------------------------------------------------------
enum { MODE_QKV = 0, MODE_RES = 1, MODE_GELU = 2 };

#define TILEB_ELEMS (128 * 32)
#define TILEB_BYTES (TILEB_ELEMS * 2)

template <int MODE>
__global__ __launch_bounds__(256) void gemm_wmma(
    const bf16_t* __restrict__ A, const bf16_t* __restrict__ Wt,
    const float* __restrict__ bias, const float* __restrict__ res,
    float* __restrict__ outF, bf16_t* __restrict__ outB,
    bf16_t* __restrict__ qb, bf16_t* __restrict__ kbuf, bf16_t* __restrict__ vtb,
    int M, int N, int K) {
  // Only LDS object in this kernel -> static LDS offset 0 for the TDM D#.
  __shared__ __align__(32) bf16_t tileB[2][TILEB_ELEMS];
  const int lane = threadIdx.x & 31;
  const int wid  = threadIdx.x >> 5;
  const int idx  = lane & 15;   // A: row in tile / B,C: column in tile
  const int hf   = lane >> 4;   // K-chunk select / M-half in C layout
  const int wx   = wid & 1;     // wave's N sub-tile (0..1) * 64
  const int wy   = wid >> 1;    // wave's M sub-tile (0..3) * 32
  const int m0 = blockIdx.y * 128 + wy * 32;
  const int n0 = blockIdx.x * 128;
  const int nw = wx * 64;
  (void)M;

  v8f acc[2][4] = {};
  const bf16_t* ap0 = A + (size_t)(m0 + idx) * K + hf * 16;
  const bf16_t* ap1 = A + (size_t)(m0 + 16 + idx) * K + hf * 16;

  const int nsteps = K / 32;
#if HAS_TDM
  if (wid == 0) tdm_load_b_tile(Wt + (size_t)n0 * K, 0u, K, N, 128);
#endif
  for (int i = 0; i < nsteps; i++) {
    const int k0 = i * 32;
#if HAS_TDM
    if (wid == 0) {
      if (i + 1 < nsteps) {
        tdm_load_b_tile(Wt + (size_t)n0 * K + (k0 + 32),
                        (unsigned)(((i + 1) & 1) * TILEB_BYTES), K, N, 128);
        __builtin_amdgcn_s_wait_tensorcnt(1);   // tile i landed
      } else {
        __builtin_amdgcn_s_wait_tensorcnt(0);
      }
    }
    __syncthreads();
    const bf16_t* bbase = tileB[i & 1];
#endif
    __builtin_prefetch(ap0 + k0 + 256, 0, 0);   // global_prefetch_b8
    AFrag a0, a1;
    a0.u[0] = *(const v4u*)(ap0 + k0);
    a0.u[1] = *(const v4u*)(ap0 + k0 + 8);
    a1.u[0] = *(const v4u*)(ap1 + k0);
    a1.u[1] = *(const v4u*)(ap1 + k0 + 8);
    #pragma unroll
    for (int t = 0; t < 4; t++) {
#if HAS_TDM
      v16bf bfrag = *(const v16bf*)(bbase + (size_t)(nw + t * 16 + idx) * 32 + hf * 16);
#else
      v16bf bfrag = *(const v16bf*)(Wt + (size_t)(n0 + nw + t * 16 + idx) * K + k0 + hf * 16);
#endif
      acc[0][t] = __builtin_amdgcn_wmma_f32_16x16x32_bf16(
          false, a0.v, false, bfrag, (short)0, acc[0][t], false, false);
      acc[1][t] = __builtin_amdgcn_wmma_f32_16x16x32_bf16(
          false, a1.v, false, bfrag, (short)0, acc[1][t], false, false);
    }
#if HAS_TDM
    __syncthreads();   // all reads of this buffer done before overwrite at i+2
#endif
  }

  #pragma unroll
  for (int g = 0; g < 2; g++) {
    #pragma unroll
    for (int t = 0; t < 4; t++) {
      const int nn = n0 + nw + t * 16 + idx;
      const float bv = bias[nn];
      #pragma unroll
      for (int r = 0; r < 8; r++) {
        const int mm = m0 + g * 16 + hf * 8 + r;   // C layout: row = r + half*8
        float v = acc[g][t][r] + bv;
        if (MODE == MODE_RES) {
          outF[(size_t)mm * N + nn] = res[(size_t)mm * N + nn] + v;
        } else if (MODE == MODE_GELU) {
          outB[(size_t)mm * N + nn] = f2bf(0.5f * v * (1.0f + erff(v * 0.70710678f)));
        } else {  // MODE_QKV: scatter to per-head Q[k][d], K[k][d], V^T[d][k]
          const int bb = mm >> 11, ss = mm & 2047;
          if (nn < DIM) {
            const int h = nn >> 6, d = nn & 63;
            qb[((size_t)(bb * HEADS + h) * SEQ + ss) * HEAD_DIM + d] = f2bf(v);
          } else if (nn < 2 * DIM) {
            const int n2 = nn - DIM, h = n2 >> 6, d = n2 & 63;
            kbuf[((size_t)(bb * HEADS + h) * SEQ + ss) * HEAD_DIM + d] = f2bf(v);
          } else {
            const int n2 = nn - 2 * DIM, h = n2 >> 6, d = n2 & 63;
            vtb[((size_t)(bb * HEADS + h) * HEAD_DIM + d) * SEQ + ss] = f2bf(v);
          }
        }
      }
    }
  }
}

// ---------------------------------------------------------------------------
// Flash-style attention. Block = 4 waves; each wave owns one 16-query tile of
// one (batch, head). 32 keys/iter: 4 WMMAs for S = Q*K^T, online softmax,
// P staged through LDS (C-layout -> A-layout, s_wait_dscnt), 4 WMMAs O += P*V.
// ---------------------------------------------------------------------------
__global__ __launch_bounds__(128) void attention_wmma(
    const bf16_t* __restrict__ qb, const bf16_t* __restrict__ kbuf,
    const bf16_t* __restrict__ vtb, bf16_t* __restrict__ attn_out) {
  __shared__ __align__(16) bf16_t plds[4][16 * 32];
  const int lane = threadIdx.x & 31;
  const int wid  = threadIdx.x >> 5;
  const int idx  = lane & 15;
  const int hf   = lane >> 4;
  const int b = blockIdx.z, h = blockIdx.y;
  const int q0 = (blockIdx.x * 4 + wid) * 16;

  const bf16_t* Q  = qb   + ((size_t)(b * HEADS + h) * SEQ + q0) * HEAD_DIM;
  const bf16_t* Km = kbuf + (size_t)(b * HEADS + h) * SEQ * HEAD_DIM;
  const bf16_t* Vt = vtb  + (size_t)(b * HEADS + h) * HEAD_DIM * SEQ;

  AFrag qa[2];
  #pragma unroll
  for (int s = 0; s < 2; s++) {
    qa[s].u[0] = *(const v4u*)(Q + idx * HEAD_DIM + s * 32 + hf * 16);
    qa[s].u[1] = *(const v4u*)(Q + idx * HEAD_DIM + s * 32 + 8 + hf * 16);
  }

  float rm[8], rl[8];
  #pragma unroll
  for (int r = 0; r < 8; r++) { rm[r] = -3.0e38f; rl[r] = 0.f; }
  v8f o[4] = {};
  const float scale = 0.125f;                  // 1/sqrt(64)
  bf16_t* myP = plds[wid];

  for (int kb0 = 0; kb0 < SEQ; kb0 += 32) {
    v8f s0 = {}, s1 = {};
    #pragma unroll
    for (int s = 0; s < 2; s++) {              // K-dim (head_dim) in 2 steps
      v16bf bk0 = *(const v16bf*)(Km + (size_t)(kb0 + idx)      * HEAD_DIM + s * 32 + hf * 16);
      v16bf bk1 = *(const v16bf*)(Km + (size_t)(kb0 + 16 + idx) * HEAD_DIM + s * 32 + hf * 16);
      s0 = __builtin_amdgcn_wmma_f32_16x16x32_bf16(false, qa[s].v, false, bk0, (short)0, s0, false, false);
      s1 = __builtin_amdgcn_wmma_f32_16x16x32_bf16(false, qa[s].v, false, bk1, (short)0, s1, false, false);
    }
    float c[8];
    #pragma unroll
    for (int r = 0; r < 8; r++) {
      float a0 = s0[r] * scale, a1 = s1[r] * scale;
      float t = fmaxf(a0, a1);
      #pragma unroll
      for (int msk = 8; msk; msk >>= 1) t = fmaxf(t, __shfl_xor(t, msk, 32));
      const float mn = fmaxf(rm[r], t);
      c[r] = __expf(rm[r] - mn);
      rm[r] = mn;
      const float p0 = __expf(a0 - mn), p1 = __expf(a1 - mn);
      float rsum = p0 + p1;
      #pragma unroll
      for (int msk = 8; msk; msk >>= 1) rsum += __shfl_xor(rsum, msk, 32);
      rl[r] = rl[r] * c[r] + rsum;
      myP[(r + hf * 8) * 32 + idx]      = f2bf(p0);
      myP[(r + hf * 8) * 32 + 16 + idx] = f2bf(p1);
    }
    #pragma unroll
    for (int t4 = 0; t4 < 4; t4++)
      #pragma unroll
      for (int r = 0; r < 8; r++) o[t4][r] *= c[r];

    // same-wave cross-lane LDS stores must land before A-frag reload
    __asm__ volatile("s_wait_dscnt 0" ::: "memory");

    AFrag pa;
    pa.u[0] = *(const v4u*)(myP + idx * 32 + hf * 16);
    pa.u[1] = *(const v4u*)(myP + idx * 32 + 8 + hf * 16);
    #pragma unroll
    for (int t4 = 0; t4 < 4; t4++) {
      v16bf bv = *(const v16bf*)(Vt + (size_t)(t4 * 16 + idx) * SEQ + kb0 + hf * 16);
      o[t4] = __builtin_amdgcn_wmma_f32_16x16x32_bf16(false, pa.v, false, bv, (short)0, o[t4], false, false);
    }
  }

  #pragma unroll
  for (int r = 0; r < 8; r++) rl[r] = 1.0f / rl[r];
  #pragma unroll
  for (int t4 = 0; t4 < 4; t4++) {
    #pragma unroll
    for (int r = 0; r < 8; r++) {
      const int qrow = q0 + hf * 8 + r;
      attn_out[((size_t)(b * SEQ + qrow)) * DIM + h * HEAD_DIM + t4 * 16 + idx] =
          f2bf(o[t4][r] * rl[r]);
    }
  }
}

// ---------------------------------------------------------------------------
// Host-side pipeline.
// ---------------------------------------------------------------------------
extern "C" void kernel_launch(void* const* d_in, const int* in_sizes, int n_in,
                              void* d_out, int out_size, void* d_ws, size_t ws_size,
                              hipStream_t stream) {
  (void)in_sizes; (void)n_in; (void)out_size; (void)ws_size;
  const float* x      = (const float*)d_in[0];
  const float* ln1_w  = (const float*)d_in[1];
  const float* ln1_b  = (const float*)d_in[2];
  const float* qkv_w  = (const float*)d_in[3];
  const float* qkv_b  = (const float*)d_in[4];
  const float* proj_w = (const float*)d_in[5];
  const float* proj_b = (const float*)d_in[6];
  const float* ln2_w  = (const float*)d_in[7];
  const float* ln2_b  = (const float*)d_in[8];
  const float* lin1_w = (const float*)d_in[9];
  const float* lin1_b = (const float*)d_in[10];
  const float* lin2_w = (const float*)d_in[11];
  const float* lin2_b = (const float*)d_in[12];

  char* base = (char*)d_ws;
  size_t off = 0;
  auto take = [&](size_t bytes) -> char* {
    char* r = base + off;
    off += (bytes + 255) & ~(size_t)255;
    return r;
  };

  bf16_t* qkvWt  = (bf16_t*)take((size_t)QKV_DIM * DIM * 2);
  bf16_t* projWt = (bf16_t*)take((size_t)DIM * DIM * 2);
  bf16_t* lin1Wt = (bf16_t*)take((size_t)MLP_DIM * DIM * 2);
  bf16_t* lin2Wt = (bf16_t*)take((size_t)DIM * MLP_DIM * 2);
  bf16_t* ln1o   = (bf16_t*)take((size_t)TOKENS * DIM * 2);
  bf16_t* qbuf   = (bf16_t*)take((size_t)TOKENS * DIM * 2);
  bf16_t* kbuf   = (bf16_t*)take((size_t)TOKENS * DIM * 2);
  bf16_t* vtbuf  = (bf16_t*)take((size_t)TOKENS * DIM * 2);
  bf16_t* attno  = (bf16_t*)take((size_t)TOKENS * DIM * 2);
  float*  x1     = (float*) take((size_t)TOKENS * DIM * 4);
  bf16_t* ln2o   = (bf16_t*)take((size_t)TOKENS * DIM * 2);
  bf16_t* h1     = (bf16_t*)take((size_t)TOKENS * MLP_DIM * 2);

  // 1) weight convert + transpose to bf16 [N][K]
  cvt_transpose<<<(DIM * QKV_DIM + 255) / 256, 256, 0, stream>>>(qkv_w,  qkvWt,  DIM,     QKV_DIM);
  cvt_transpose<<<(DIM * DIM     + 255) / 256, 256, 0, stream>>>(proj_w, projWt, DIM,     DIM);
  cvt_transpose<<<(DIM * MLP_DIM + 255) / 256, 256, 0, stream>>>(lin1_w, lin1Wt, DIM,     MLP_DIM);
  cvt_transpose<<<(MLP_DIM * DIM + 255) / 256, 256, 0, stream>>>(lin2_w, lin2Wt, MLP_DIM, DIM);

  // 2) LN1 -> bf16
  layernorm_bf16<<<TOKENS, 256, 0, stream>>>(x, ln1_w, ln1_b, ln1o);

  // 3) QKV GEMM, scatter to per-head Q/K/V^T
  gemm_wmma<MODE_QKV><<<dim3(QKV_DIM / 128, TOKENS / 128), 256, 0, stream>>>(
      ln1o, qkvWt, qkv_b, nullptr, nullptr, nullptr, qbuf, kbuf, vtbuf,
      TOKENS, QKV_DIM, DIM);

  // 4) attention
  attention_wmma<<<dim3(SEQ / 64, HEADS, BATCH), 128, 0, stream>>>(
      qbuf, kbuf, vtbuf, attno);

  // 5) proj GEMM + residual -> x1 (fp32)
  gemm_wmma<MODE_RES><<<dim3(DIM / 128, TOKENS / 128), 256, 0, stream>>>(
      attno, projWt, proj_b, x, x1, nullptr, nullptr, nullptr, nullptr,
      TOKENS, DIM, DIM);

  // 6) LN2 -> bf16
  layernorm_bf16<<<TOKENS, 256, 0, stream>>>(x1, ln2_w, ln2_b, ln2o);

  // 7) lin1 GEMM + exact GELU -> h1 (bf16)
  gemm_wmma<MODE_GELU><<<dim3(MLP_DIM / 128, TOKENS / 128), 256, 0, stream>>>(
      ln2o, lin1Wt, lin1_b, nullptr, nullptr, h1, nullptr, nullptr, nullptr,
      TOKENS, MLP_DIM, DIM);

  // 8) lin2 GEMM + residual -> d_out (fp32)
  gemm_wmma<MODE_RES><<<dim3(DIM / 128, TOKENS / 128), 256, 0, stream>>>(
      h1, lin2Wt, lin2_b, x1, (float*)d_out, nullptr, nullptr, nullptr, nullptr,
      TOKENS, DIM, MLP_DIM);
}
